// SAGNN_41008347742504
// MI455X (gfx1250) — matrix operand order repaired and checked by reading, not compiled, not run on hardware
//
#include <hip/hip_runtime.h>
#include <hip/hip_fp16.h>

typedef __attribute__((ext_vector_type(16))) _Float16 v16h;
typedef __attribute__((ext_vector_type(8)))  _Float16 h8;
typedef __attribute__((ext_vector_type(8)))  float    v8f;
typedef __attribute__((ext_vector_type(4)))  float    f4;

#define NEG_SLOPE 0.2f

__device__ __forceinline__ float sigmoidf_(float x) { return 1.0f / (1.0f + expf(-x)); }

#define WMMA_F16(a, b, c) \
    __builtin_amdgcn_wmma_f32_16x16x32_f16(false, (a), false, (b), (short)0, (c), false, false)

// ---- WMMA fragment loaders (layouts per CDNA5 ISA 7.12.2, f16 16x16x32) ----
// A (f32 source, converted inline): lane m = lane&15 (row ptr passed in);
// elements 0..7 -> k = kk+hi*8+e, elements 8..15 -> k = kk+16+hi*8+(e-8).
__device__ __forceinline__ v16h load_a_full(const float* __restrict__ Arow, int kk, int hi) {
    const f4* p = (const f4*)(Arow + kk + hi * 8);
    const f4* q = (const f4*)(Arow + kk + 16 + hi * 8);
    f4 v0 = p[0], v1 = p[1], v2 = q[0], v3 = q[1];
    v16h a;
#pragma unroll
    for (int e = 0; e < 4; ++e) {
        a[e]      = (_Float16)v0[e];
        a[e + 4]  = (_Float16)v1[e];
        a[e + 8]  = (_Float16)v2[e];
        a[e + 12] = (_Float16)v3[e];
    }
    return a;
}

// K=100 tail (kk=96): valid k = 96..99 only (hi==0, e<4). One in-bounds b128 + cndmask.
__device__ __forceinline__ v16h load_a_tail100(const float* __restrict__ Arow, int hi) {
    f4 v = *(const f4*)(Arow + 96);
    v16h a;
#pragma unroll
    for (int e = 0; e < 16; ++e) a[e] = (_Float16)0.f;
#pragma unroll
    for (int e = 0; e < 4; ++e) a[e] = (hi == 0) ? (_Float16)v[e] : (_Float16)0.f;
    return a;
}

// B from pre-converted f16 row-major Wh[Nout][Kp] (K zero-padded to mult of 32).
// lane n = col0+lane&15 (row ptr passed in); elements 0..7 -> k = kk+hi*8+e, 8..15 -> k = kk+16+hi*8+(e-8).
// two runs of 8 halves = 16B each -> two b128 loads, zero VALU.
__device__ __forceinline__ v16h load_bh(const _Float16* __restrict__ Wrow, int kk, int hi) {
    const h8* p = (const h8*)(Wrow + kk + hi * 8);
    const h8* q = (const h8*)(Wrow + kk + 16 + hi * 8);
    h8 v0 = p[0], v1 = q[0];
    v16h b;
#pragma unroll
    for (int e = 0; e < 8; ++e) { b[e] = v0[e]; b[e + 8] = v1[e]; }
    return b;
}

// ---- weight pre-convert: f32 [rows][K] -> f16 [rows][Kp], zero-padded ----
__global__ void k_cvt_w(const float* __restrict__ W, _Float16* __restrict__ Wh,
                        int rows, int K, int Kp) {
    int t = blockIdx.x * blockDim.x + threadIdx.x;
    if (t >= rows * Kp) return;
    int r = t / Kp, k = t - r * Kp;
    Wh[t] = (k < K) ? (_Float16)W[(size_t)r * K + k] : (_Float16)0.f;
}

// ---- generic GEMM: out[M][Nout] = A[M][K] @ W[Nout][K]^T.  M%16==0, Nout%16==0. ----
template <int K>
__global__ void k_wmma_gemm(const float* __restrict__ A, const _Float16* __restrict__ Wh,
                            float* __restrict__ out, int Nout, int Kp) {
    constexpr int NC = (K + 31) / 32;
    int row0 = blockIdx.x * 16, col0 = blockIdx.y * 16;
    int lane = threadIdx.x, hi = lane >> 4, c16 = lane & 15;
    const float* Arow = A + (size_t)(row0 + c16) * K;
    const _Float16* Wrow = Wh + (size_t)(col0 + c16) * Kp;
    v16h a[NC];
#pragma unroll
    for (int c = 0; c < K / 32; ++c) a[c] = load_a_full(Arow, c * 32, hi);
    if constexpr (K % 32 != 0) a[NC - 1] = load_a_tail100(Arow, hi);
    v8f acc = {};
#pragma unroll
    for (int c = 0; c < NC; ++c)
        acc = WMMA_F16(a[c], load_bh(Wrow, c * 32, hi), acc);
    int c = col0 + c16;
#pragma unroll
    for (int r = 0; r < 8; ++r)
        out[(size_t)(row0 + r + hi * 8) * Nout + c] = acc[r];
}

// ---- leaf stage: per CFG node, 15x100 @ 100x192 via WMMA, fused LSTM gates ----
// A preloaded once (4 frags), tn outer / kk inner keeps one B frag in flight.
__global__ void k_leaf(const float* __restrict__ ast_x, const float* __restrict__ c_init,
                       const _Float16* __restrict__ Wiou_h, const float* __restrict__ b_iou,
                       float* __restrict__ h_leaf, float* __restrict__ c_leaf) {
    int n = blockIdx.x, lane = threadIdx.x;
    int hi = lane >> 4, c16 = lane & 15;
    int mrow = (c16 < 15) ? c16 : 14;           // clamp pad row (C row 15 never stored)
    const float* Arow = ast_x + ((size_t)n * 16 + 1 + mrow) * 100;
    v16h a[4];
#pragma unroll
    for (int c = 0; c < 3; ++c) a[c] = load_a_full(Arow, c * 32, hi);
    a[3] = load_a_tail100(Arow, hi);
    const _Float16* Wbase = Wiou_h + (size_t)c16 * 128;
    v8f acc[12] = {};
#pragma unroll
    for (int tn = 0; tn < 12; ++tn) {
        const _Float16* Wrow = Wbase + tn * (16 * 128);
#pragma unroll
        for (int c = 0; c < 4; ++c)
            acc[tn] = WMMA_F16(a[c], load_bh(Wrow, c * 32, hi), acc[tn]);
    }
#pragma unroll
    for (int ti = 0; ti < 4; ++ti) {
        int col = ti * 16 + c16;
        float bi = b_iou[col], bo = b_iou[col + 64], bu = b_iou[col + 128];
#pragma unroll
        for (int r = 0; r < 8; ++r) {
            int t = r + hi * 8;                 // leaf index 0..14 (15 = pad)
            if (r < 7 || hi == 0) {             // only (r==7,hi==1) is padded
                float iv = acc[ti][r] + bi;
                float ov = acc[ti + 4][r] + bo;
                float uv = acc[ti + 8][r] + bu;
                float c0 = c_init[((size_t)n * 16 + 1 + t) * 64 + col];
                float cv = sigmoidf_(iv) * tanhf(uv) + c0;
                float hv = sigmoidf_(ov) * tanhf(cv);
                size_t idx = ((size_t)n * 15 + t) * 64 + col;
                c_leaf[idx] = cv;
                h_leaf[idx] = hv;
            }
        }
    }
}

// ---- child reduction: f = sigmoid(h_leaf @ U_f_W^T + b); c_red = sum f*c; h_tild = sum h ----
__global__ void k_reduce(const float* __restrict__ h_leaf, const float* __restrict__ c_leaf,
                         const _Float16* __restrict__ Ufw_h, const float* __restrict__ U_f_b,
                         float* __restrict__ h_tild, float* __restrict__ c_red) {
    int n = blockIdx.x, lane = threadIdx.x;
    int hi = lane >> 4, c16 = lane & 15;
    int mrow = (c16 < 15) ? c16 : 14;
    const float* Arow = h_leaf + ((size_t)n * 15 + mrow) * 64;
    v16h a[2];
#pragma unroll
    for (int c = 0; c < 2; ++c) a[c] = load_a_full(Arow, c * 32, hi);
    const _Float16* Wbase = Ufw_h + (size_t)c16 * 64;
    v8f acc[4] = {};
#pragma unroll
    for (int tn = 0; tn < 4; ++tn) {
        const _Float16* Wrow = Wbase + tn * (16 * 64);
#pragma unroll
        for (int c = 0; c < 2; ++c)
            acc[tn] = WMMA_F16(a[c], load_bh(Wrow, c * 32, hi), acc[tn]);
    }
#pragma unroll
    for (int ti = 0; ti < 4; ++ti) {
        int col = ti * 16 + c16;
        float bf = U_f_b[col];
        float cs = 0.f, hs = 0.f;
#pragma unroll
        for (int r = 0; r < 8; ++r) {
            int t = r + hi * 8;
            if (r < 7 || hi == 0) {
                float f = sigmoidf_(acc[ti][r] + bf);
                size_t idx = ((size_t)n * 15 + t) * 64 + col;
                cs += f * c_leaf[idx];
                hs += h_leaf[idx];
            }
        }
        cs += __shfl_xor(cs, 16, 32);   // combine t=0..7 with t=8..15 lane halves
        hs += __shfl_xor(hs, 16, 32);
        if (hi == 0) {
            c_red[(size_t)n * 64 + col]  = cs;
            h_tild[(size_t)n * 64 + col] = hs;
        }
    }
}

// ---- root: iou_r = h_tild @ U_iou^T + b_iou, fused gates -> h_root (N%16==0) ----
__global__ void k_root(const float* __restrict__ h_tild, const float* __restrict__ c_red,
                       const _Float16* __restrict__ Uiou_h, const float* __restrict__ b_iou,
                       float* __restrict__ h_root) {
    int row0 = blockIdx.x * 16, lane = threadIdx.x;
    int hi = lane >> 4, c16 = lane & 15;
    const float* Arow = h_tild + (size_t)(row0 + c16) * 64;
    v16h a[2];
#pragma unroll
    for (int c = 0; c < 2; ++c) a[c] = load_a_full(Arow, c * 32, hi);
    const _Float16* Wbase = Uiou_h + (size_t)c16 * 64;
    v8f acc[12] = {};
#pragma unroll
    for (int tn = 0; tn < 12; ++tn) {
        const _Float16* Wrow = Wbase + tn * (16 * 64);
#pragma unroll
        for (int c = 0; c < 2; ++c)
            acc[tn] = WMMA_F16(a[c], load_bh(Wrow, c * 32, hi), acc[tn]);
    }
#pragma unroll
    for (int ti = 0; ti < 4; ++ti) {
        int col = ti * 16 + c16;
        float bi = b_iou[col], bo = b_iou[col + 64], bu = b_iou[col + 128];
#pragma unroll
        for (int r = 0; r < 8; ++r) {
            int m = row0 + r + hi * 8;
            float iv = acc[ti][r] + bi, ov = acc[ti + 4][r] + bo, uv = acc[ti + 8][r] + bu;
            float cv = sigmoidf_(iv) * tanhf(uv) + c_red[(size_t)m * 64 + col];
            h_root[(size_t)m * 64 + col] = sigmoidf_(ov) * tanhf(cv);
        }
    }
}

// ---- attention scalars ----
__global__ void k_attn(const float* __restrict__ fsrc, const float* __restrict__ attn_l,
                       const float* __restrict__ attn_r, float* __restrict__ el,
                       float* __restrict__ er, int N4) {
    int t = blockIdx.x * blockDim.x + threadIdx.x;
    if (t >= N4) return;
    int h = t & 3;
    const float* row = fsrc + (size_t)(t >> 2) * 128 + h * 32;
    float sl = 0.f, sr = 0.f;
#pragma unroll 8
    for (int o = 0; o < 32; ++o) {
        float v = row[o];
        sl += v * attn_l[h * 32 + o];
        sr += v * attn_r[h * 32 + o];
    }
    el[t] = sl;
    er[t] = sr;
}

// order-preserving float->uint key for atomic segment-max (key 0 == -inf floor)
__device__ __forceinline__ unsigned fkey(float f) {
    unsigned u = __float_as_uint(f);
    return (u & 0x80000000u) ? ~u : (u | 0x80000000u);
}
__device__ __forceinline__ float funkey(unsigned k) {
    unsigned u = (k & 0x80000000u) ? (k & 0x7fffffffu) : ~k;
    return __uint_as_float(u);
}

__global__ void k_edge1(const float* __restrict__ el, const float* __restrict__ er,
                        const int* __restrict__ esrc, const int* __restrict__ edst,
                        float* __restrict__ e_buf, unsigned* __restrict__ emax, int E4) {
    int t = blockIdx.x * blockDim.x + threadIdx.x;
    if (t >= E4) return;
    int e = t >> 2, h = t & 3;
    int s = esrc[e], d = edst[e];
    float v = el[s * 4 + h] + er[d * 4 + h];
    v = v > 0.f ? v : NEG_SLOPE * v;
    e_buf[t] = v;
    atomicMax(&emax[d * 4 + h], fkey(v));
}

__global__ void k_edge2(float* __restrict__ e_buf, const unsigned* __restrict__ emax,
                        float* __restrict__ esum, const int* __restrict__ edst, int E4) {
    int t = blockIdx.x * blockDim.x + threadIdx.x;
    if (t >= E4) return;
    int e = t >> 2, h = t & 3;
    int d = edst[e];
    float ez = expf(e_buf[t] - funkey(emax[d * 4 + h]));
    e_buf[t] = ez;
    atomicAdd(&esum[d * 4 + h], ez);
}

// m = h_root[src] * a; segment_sum by dst. One block per edge, 256 lanes = 4 heads x 64 ch.
__global__ void k_scatter(const float* __restrict__ e_buf, const float* __restrict__ esum,
                          const float* __restrict__ h_root, const int* __restrict__ esrc,
                          const int* __restrict__ edst, float* __restrict__ agg) {
    int e = blockIdx.x;
    int rem = threadIdx.x;
    int h = rem >> 6, c = rem & 63;
    int s = esrc[e], d = edst[e];
    float a = e_buf[e * 4 + h] / esum[d * 4 + h];
    atomicAdd(&agg[((size_t)d * 4 + h) * 64 + c], h_root[(size_t)s * 64 + c] * a);
}

// legacy dim=0 softmax: one block per (head,out) column, stride 128 over N rows
__global__ void k_softmax0(const float* __restrict__ rst_pre, const float* __restrict__ bias_g,
                           float* __restrict__ rst, int N) {
    int j = blockIdx.x;            // 0..127
    float b = bias_g[j];
    __shared__ float red[256];
    int tid = threadIdx.x;
    float mx = -3.0e38f;
    for (int n = tid; n < N; n += 256) mx = fmaxf(mx, rst_pre[(size_t)n * 128 + j] + b);
    red[tid] = mx; __syncthreads();
    for (int s = 128; s > 0; s >>= 1) { if (tid < s) red[tid] = fmaxf(red[tid], red[tid + s]); __syncthreads(); }
    mx = red[0]; __syncthreads();
    float sm = 0.f;
    for (int n = tid; n < N; n += 256) sm += expf(rst_pre[(size_t)n * 128 + j] + b - mx);
    red[tid] = sm; __syncthreads();
    for (int s = 128; s > 0; s >>= 1) { if (tid < s) red[tid] += red[tid + s]; __syncthreads(); }
    float inv = 1.f / red[0];
    for (int n = tid; n < N; n += 256)
        rst[(size_t)n * 128 + j] = expf(rst_pre[(size_t)n * 128 + j] + b - mx) * inv;
}

// tiny classifier: [*,32] -> 16 -> 2
__global__ void k_cls(const float* __restrict__ rst, const float* __restrict__ W1,
                      const float* __restrict__ b1, const float* __restrict__ W2,
                      const float* __restrict__ b2, float* __restrict__ out, int R) {
    int t = blockIdx.x * blockDim.x + threadIdx.x;
    if (t >= R) return;
    const float* row = rst + (size_t)t * 32;
    float mid[16];
#pragma unroll
    for (int mi = 0; mi < 16; ++mi) {
        float s = b1[mi];
        for (int o = 0; o < 32; ++o) s += row[o] * W1[mi * 32 + o];
        mid[mi] = s;
    }
#pragma unroll
    for (int k = 0; k < 2; ++k) {
        float s = b2[k];
        for (int mi = 0; mi < 16; ++mi) s += mid[mi] * W2[k * 16 + mi];
        out[(size_t)t * 2 + k] = s;
    }
}

extern "C" void kernel_launch(void* const* d_in, const int* in_sizes, int n_in,
                              void* d_out, int out_size, void* d_ws, size_t ws_size,
                              hipStream_t stream) {
    const float* ast_x    = (const float*)d_in[0];
    const float* c_init   = (const float*)d_in[1];
    const float* cfg_type = (const float*)d_in[2];
    const float* W_iou    = (const float*)d_in[3];
    const float* b_iou    = (const float*)d_in[4];
    const float* U_iou    = (const float*)d_in[5];
    const float* U_f_W    = (const float*)d_in[6];
    const float* U_f_b    = (const float*)d_in[7];
    const float* Wt       = (const float*)d_in[8];
    const float* Wfc      = (const float*)d_in[9];
    const float* attn_l   = (const float*)d_in[10];
    const float* attn_r   = (const float*)d_in[11];
    const float* W_out    = (const float*)d_in[12];
    const float* bias_g   = (const float*)d_in[13];
    const float* W1       = (const float*)d_in[14];
    const float* b1       = (const float*)d_in[15];
    const float* W2       = (const float*)d_in[16];
    const float* b2       = (const float*)d_in[17];
    const int*   esrc     = (const int*)d_in[18];
    const int*   edst     = (const int*)d_in[19];

    const int N = in_sizes[2] / 100;     // 20000 (multiple of 16)
    const int E = in_sizes[18];          // 640000

    // ---- workspace layout (floats) ----
    float* ws = (float*)d_ws;
    size_t off = 0;
    float* h_leaf  = ws + off; off += (size_t)N * 15 * 64;
    float* c_leaf  = ws + off; off += (size_t)N * 15 * 64;
    float* h_tild  = ws + off; off += (size_t)N * 64;
    float* c_red   = ws + off; off += (size_t)N * 64;
    float* h_root  = ws + off; off += (size_t)N * 64;
    float* feat    = ws + off; off += (size_t)N * 64;
    float* fsrc    = ws + off; off += (size_t)N * 128;
    float* el      = ws + off; off += (size_t)N * 4;
    float* er      = ws + off; off += (size_t)N * 4;
    float* e_buf   = ws + off; off += (size_t)E * 4;
    unsigned* emax = (unsigned*)(ws + off); off += (size_t)N * 4;   // contiguous with esum+agg
    float* esum    = ws + off; off += (size_t)N * 4;
    float* agg     = ws + off; off += (size_t)N * 256;
    float* rst_pre = ws + off; off += (size_t)N * 128;
    float* rst     = ws + off; off += (size_t)N * 128;
    // f16 pre-converted weights (K zero-padded to mult of 32)
    _Float16* Wiou_h = (_Float16*)(ws + off); off += (192 * 128) / 2;
    _Float16* Uiou_h = (_Float16*)(ws + off); off += (192 * 64) / 2;
    _Float16* Ufw_h  = (_Float16*)(ws + off); off += (64 * 64) / 2;
    _Float16* Wt_h   = (_Float16*)(ws + off); off += (64 * 128) / 2;
    _Float16* Wfc_h  = (_Float16*)(ws + off); off += (128 * 64) / 2;
    _Float16* Wout_h = (_Float16*)(ws + off); off += (32 * 64) / 2;

    // zero emax (key 0 == -inf), esum, agg in one contiguous memset (capturable)
    hipMemsetAsync(emax, 0, (size_t)(N * 4 + N * 4 + N * 256) * sizeof(float), stream);

    // weight pre-conversion (tiny, one-shot per call)
    k_cvt_w<<<(192 * 128 + 255) / 256, 256, 0, stream>>>(W_iou, Wiou_h, 192, 100, 128);
    k_cvt_w<<<(192 * 64 + 255) / 256, 256, 0, stream>>>(U_iou, Uiou_h, 192, 64, 64);
    k_cvt_w<<<(64 * 64 + 255) / 256, 256, 0, stream>>>(U_f_W, Ufw_h, 64, 64, 64);
    k_cvt_w<<<(64 * 128 + 255) / 256, 256, 0, stream>>>(Wt, Wt_h, 64, 100, 128);
    k_cvt_w<<<(128 * 64 + 255) / 256, 256, 0, stream>>>(Wfc, Wfc_h, 128, 64, 64);
    k_cvt_w<<<(32 * 64 + 255) / 256, 256, 0, stream>>>(W_out, Wout_h, 32, 64, 64);

    // Tree-LSTM
    k_leaf<<<N, 32, 0, stream>>>(ast_x, c_init, Wiou_h, b_iou, h_leaf, c_leaf);
    k_reduce<<<N, 32, 0, stream>>>(h_leaf, c_leaf, Ufw_h, U_f_b, h_tild, c_red);
    k_root<<<N / 16, 32, 0, stream>>>(h_tild, c_red, Uiou_h, b_iou, h_root);

    // TGATConv
    k_wmma_gemm<100><<<dim3(N / 16, 4), 32, 0, stream>>>(cfg_type, Wt_h, feat, 64, 128);
    k_wmma_gemm<64><<<dim3(N / 16, 8), 32, 0, stream>>>(feat, Wfc_h, fsrc, 128, 64);
    k_attn<<<(N * 4 + 255) / 256, 256, 0, stream>>>(fsrc, attn_l, attn_r, el, er, N * 4);
    k_edge1<<<(E * 4 + 255) / 256, 256, 0, stream>>>(el, er, esrc, edst, e_buf, emax, E * 4);
    k_edge2<<<(E * 4 + 255) / 256, 256, 0, stream>>>(e_buf, emax, esum, edst, E * 4);
    k_scatter<<<E, 256, 0, stream>>>(e_buf, esum, h_root, esrc, edst, agg);
    k_wmma_gemm<64><<<dim3(N * 4 / 16, 2), 32, 0, stream>>>(agg, Wout_h, rst_pre, 32, 64);
    k_softmax0<<<128, 256, 0, stream>>>(rst_pre, bias_g, rst, N);
    k_cls<<<(N * 4 + 255) / 256, 256, 0, stream>>>(rst, W1, b1, W2, b2, (float*)d_out, N * 4);
}
